// FeaturizeProtein_84172769068216
// MI455X (gfx1250) — matrix operand order repaired and checked by reading, not compiled, not run on hardware
//
#include <hip/hip_runtime.h>

typedef __attribute__((ext_vector_type(16))) _Float16 v16h;
typedef __attribute__((ext_vector_type(8)))  float    v8f;

#define ZB      2
#define NN      4096
#define KNNK    30
#define EDIM    256
#define DMODEL  128
#define NEDGE   (ZB*NN*KNNK)        // 245760
// exp(-dd^2 / SPREAD^2) = exp2(dd^2 * (-1/SPREAD^2 * log2(e))), SPREAD = 1.25
#define NEG_INV_SPREAD2_LOG2E  (-0.9233248026110401f)

// ---------------------------------------------------------------------------
// Kernel 0: W (256x128 f32 row-major) -> f16, B-fragment order for
// v_wmma_f32_16x16x32_f16.  B (32K x 16N): lanes 0-15 hold col N=lane,
// K=0..7,16..23 (elem p); lanes 16-31 hold N=lane-16, K=8..15,24..31.
// Stored as [c][n][lane][p] so a lane's 16 halves are 32 contiguous bytes.
// ---------------------------------------------------------------------------
__global__ __launch_bounds__(256) void prep_w_kernel(const float* __restrict__ W,
                                                     _Float16* __restrict__ Wf) {
    int o    = blockIdx.x * 256 + threadIdx.x;   // 0..32767
    int p    =  o        & 15;
    int lane = (o >> 4)  & 31;
    int n    = (o >> 9)  & 7;
    int c    =  o >> 12;
    int h    = lane >> 4;
    int k    = 32*c + 8*h + (p & 7) + 16*(p >> 3);
    int col  = 16*n + (lane & 15);
    Wf[o] = (_Float16)W[k*DMODEL + col];
}

// ---------------------------------------------------------------------------
// Kernel 1: KNN.  One workgroup per (z,i).  Each thread owns 16 candidates
// (j = t + 256r), sorts them ascending in registers (bitonic, static
// indices), spills the sorted list to LDS and keeps the head in red[t].
// Each of the 30 extractions: wave0 min-reduces red[256] (ds + shfl, no
// intra-wave barriers), broadcasts; owner thread pops its next sorted key.
// 2 barriers per extraction.  Keys (f32bits(d2)<<32)|j are monotonic for
// d2>=0 and break ties toward lowest j, matching jax top_k order.
// ---------------------------------------------------------------------------
__global__ __launch_bounds__(256) void knn_kernel(const float* __restrict__ C,
                                                  const unsigned char* __restrict__ nmask,
                                                  int* __restrict__ kidx,
                                                  unsigned char* __restrict__ emask) {
    __shared__ unsigned long long skeys[256*16];   // 32 KB sorted candidate lists
    __shared__ unsigned long long red[256];        // current head per thread
    __shared__ unsigned long long bcast;
    __shared__ unsigned char     lmask[NN];        // node_mask[z,:]

    const int bid  = blockIdx.x;                   // z*N + i
    const int z    = bid >> 12;
    const int i    = bid & (NN - 1);
    const int t    = threadIdx.x;
    const int lane = t & 31;
    const int wv   = t >> 5;
    const float inf = __uint_as_float(0x7f800000u);

    ((uint4*)lmask)[t] = ((const uint4*)(nmask + z*NN))[t];   // 16 B/thread

    const size_t cbase = (size_t)z * NN * 12;      // C[z,:,:,:], Ca at atom 1
    const float xi = C[cbase + (size_t)i*12 + 3];
    const float yi = C[cbase + (size_t)i*12 + 4];
    const float zi = C[cbase + (size_t)i*12 + 5];
    const int maskI = nmask[z*NN + i];

    unsigned long long a[16];
    #pragma unroll
    for (int r = 0; r < 16; ++r) {
        int j = t + 256*r;
        float dx = C[cbase + (size_t)j*12 + 3] - xi;
        float dy = C[cbase + (size_t)j*12 + 4] - yi;
        float dz = C[cbase + (size_t)j*12 + 5] - zi;
        float sq = dx*dx + dy*dy + dz*dz;
        if (sq == 0.0f || maskI) sq = inf;         // self/identical or masked row
        a[r] = ((unsigned long long)__float_as_uint(sq) << 32) | (unsigned int)j;
    }

    // bitonic sort, ascending, fully unrolled (all register indices static)
    #pragma unroll
    for (int kk = 2; kk <= 16; kk <<= 1) {
        #pragma unroll
        for (int jj = kk >> 1; jj > 0; jj >>= 1) {
            #pragma unroll
            for (int x = 0; x < 16; ++x) {
                const int l = x ^ jj;
                if (l > x) {
                    const bool up = ((x & kk) == 0);
                    unsigned long long lo = (a[x] < a[l]) ? a[x] : a[l];
                    unsigned long long hi = (a[x] < a[l]) ? a[l] : a[x];
                    a[x] = up ? lo : hi;
                    a[l] = up ? hi : lo;
                }
            }
        }
    }
    #pragma unroll
    for (int r = 0; r < 16; ++r) skeys[t*16 + r] = a[r];

    unsigned long long cand = a[0];
    int ptr = 0;
    red[t] = cand;
    __syncthreads();

    for (int k = 0; k < KNNK; ++k) {
        if (wv == 0) {                              // wave0: global argmin of red[]
            unsigned long long b = red[lane];
            #pragma unroll
            for (int r = 1; r < 8; ++r) {
                unsigned long long v = red[lane + 32*r];
                b = (v < b) ? v : b;
            }
            #pragma unroll
            for (int s = 16; s > 0; s >>= 1) {
                unsigned long long o = __shfl_xor(b, s, 32);
                b = (o < b) ? o : b;
            }
            if (lane == 0) bcast = b;
        }
        __syncthreads();
        const unsigned long long key0 = bcast;
        if (key0 == cand) {                         // unique keys -> single owner
            ++ptr;
            cand = (ptr < 16) ? skeys[t*16 + ptr] : ~0ull;
            red[t] = cand;
        }
        if (t == 0) {
            int jstar = (int)(key0 & 0xffffffffu);
            float d2 = __uint_as_float((unsigned int)(key0 >> 32));
            int maskJ = lmask[jstar];
            bool valid = (!maskI) && (!maskJ) && (d2 != 0.0f) && (d2 < 144.0f);
            kidx[bid*KNNK + k] = valid ? jstar : i;
            emask[bid*KNNK + k] = valid ? (unsigned char)1 : (unsigned char)0;
        }
        __syncthreads();
    }
}

// ---------------------------------------------------------------------------
// Kernel 2: edge features + LN + GEMM via v_wmma_f32_16x16x32_f16.
// 8 waves/block; wave = one 16-edge tile.  Lane (m=lane&15, h=lane>>4)
// computes edge m's features at K = 32c + (p&7) + 8h + 16(p>>3): exactly the
// 16-bit A-fragment layout, so A is built in registers.  The A fragment for
// chunk c is (re)built immediately before its 8 WMMAs so only one v16h is
// live at a time (keeps VGPRs < 256 for occupancy).  W fragments staged in
// 64 KB LDS per block.
// ---------------------------------------------------------------------------
__global__ __launch_bounds__(256) void edge_kernel(const float* __restrict__ C,
                                                   const int* __restrict__ kidx,
                                                   const float* __restrict__ centers,
                                                   const float* __restrict__ gamma,
                                                   const float* __restrict__ beta,
                                                   const float* __restrict__ bias,
                                                   const _Float16* __restrict__ Wf,
                                                   float* __restrict__ Eout) {
    __shared__ __align__(32) uint4 ldsW[4096];       // 64 KB of B fragments
    __shared__ float ldsG[EDIM], ldsBt[EDIM], ldsCent[16], ldsBias[DMODEL];

    const int t = threadIdx.x;
    const uint4* wsrc = (const uint4*)Wf;
    #pragma unroll
    for (int r = 0; r < 16; ++r) ldsW[t + 256*r] = wsrc[t + 256*r];
    ldsG[t]  = gamma[t];
    ldsBt[t] = beta[t];
    if (t < DMODEL) ldsBias[t] = bias[t];
    if (t < 16)     ldsCent[t] = centers[t];
    __syncthreads();

    const int lane = t & 31;
    const int wave = t >> 5;
    const int tile = blockIdx.x * 8 + wave;          // 15360 tiles total
    const int m = lane & 15;
    const int h = lane >> 4;
    const int eg = tile*16 + m;                      // global edge index

    const int z = eg / (NN*KNNK);
    const int rem = eg - z*(NN*KNNK);
    const int i = rem / KNNK;
    const int j = kidx[eg];                          // invalid edges: j == i

    float Pi[12], Pj[12];
    {
        const float4* pi4 = (const float4*)(C + (size_t)(z*NN + i)*12);
        const float4* pj4 = (const float4*)(C + (size_t)(z*NN + j)*12);
        float4 a0 = pi4[0], a1 = pi4[1], a2 = pi4[2];
        Pi[0]=a0.x; Pi[1]=a0.y; Pi[2]=a0.z;  Pi[3]=a0.w;
        Pi[4]=a1.x; Pi[5]=a1.y; Pi[6]=a1.z;  Pi[7]=a1.w;
        Pi[8]=a2.x; Pi[9]=a2.y; Pi[10]=a2.z; Pi[11]=a2.w;
        float4 b0 = pj4[0], b1 = pj4[1], b2 = pj4[2];
        Pj[0]=b0.x; Pj[1]=b0.y; Pj[2]=b0.z;  Pj[3]=b0.w;
        Pj[4]=b1.x; Pj[5]=b1.y; Pj[6]=b1.z;  Pj[7]=b1.w;
        Pj[8]=b2.x; Pj[9]=b2.y; Pj[10]=b2.z; Pj[11]=b2.w;
    }
    float dist[16];
    #pragma unroll
    for (int a1i = 0; a1i < 4; ++a1i)
        #pragma unroll
        for (int a2i = 0; a2i < 4; ++a2i) {
            float dx = Pi[a1i*3+0] - Pj[a2i*3+0];
            float dy = Pi[a1i*3+1] - Pj[a2i*3+1];
            float dz = Pi[a1i*3+2] - Pj[a2i*3+2];
            float sq = dx*dx + dy*dy + dz*dz;
            dist[a1i*4+a2i] = (sq > 0.0f) ? sqrtf(sq) : 0.0f;  // _safe_dist
        }
    float centv[16];
    #pragma unroll
    for (int q = 0; q < 16; ++q) centv[q] = ldsCent[q];

    // pass 1: this lane's 128 features -> partial mean / sumsq.
    // e0 and e0+8 share (a1,a2); only the RBF center index depends on h.
    float sum = 0.f, ss = 0.f;
    #pragma unroll
    for (int c = 0; c < 8; ++c)
        #pragma unroll
        for (int p = 0; p < 16; ++p) {
            const int e0 = 32*c + (p&7) + 16*(p>>3);
            const int A1 = e0 >> 6, A2 = (e0 >> 4) & 3;
            float ctr = h ? centv[(p&7)+8] : centv[p&7];
            float dd  = dist[A1*4+A2] - ctr;
            float v   = exp2f(dd*dd * NEG_INV_SPREAD2_LOG2E);
            sum += v; ss += v*v;
        }
    sum += __shfl_xor(sum, 16, 32);                  // combine with lane^16 (wave32)
    ss  += __shfl_xor(ss , 16, 32);
    const float mu   = sum * (1.0f/EDIM);
    const float var  = ss  * (1.0f/EDIM) - mu*mu;
    const float rstd = rsqrtf(var + 1e-5f);

    // GEMM: D(16x128) = A(16x256) x W(256x128), K in 8 chunks of 32.
    // A fragment for chunk c is rebuilt (recompute + LN + f16 pack) right
    // before its 8 WMMAs -> short v16h lifetime, low register pressure.
    v8f acc[8] = {};
    #pragma unroll
    for (int c = 0; c < 8; ++c) {
        v16h af;
        #pragma unroll
        for (int p = 0; p < 16; ++p) {
            const int e0 = 32*c + (p&7) + 16*(p>>3);
            const int A1 = e0 >> 6, A2 = (e0 >> 4) & 3;
            const int e  = e0 + 8*h;
            float ctr = h ? centv[(p&7)+8] : centv[p&7];
            float dd  = dist[A1*4+A2] - ctr;
            float v   = exp2f(dd*dd * NEG_INV_SPREAD2_LOG2E);
            float y   = (v - mu) * rstd * ldsG[e] + ldsBt[e];
            af[p]     = (_Float16)y;
        }
        #pragma unroll
        for (int n = 0; n < 8; ++n) {
            union { uint4 u[2]; v16h v; } bf;
            const int off = ((c*8 + n)*32 + lane)*2;
            bf.u[0] = ldsW[off];
            bf.u[1] = ldsW[off+1];
            acc[n] = __builtin_amdgcn_wmma_f32_16x16x32_f16(
                         false, af, false, bf.v, (short)0, acc[n], false, false);
        }
    }

    // store: C/D layout -> VGPR r, lanes0-15: row M=r; lanes16-31: row M=r+8
    #pragma unroll
    for (int n = 0; n < 8; ++n) {
        const int col = 16*n + m;
        const float bc = ldsBias[col];
        #pragma unroll
        for (int r = 0; r < 8; ++r) {
            const size_t row = (size_t)tile*16 + r + 8*h;
            Eout[row*DMODEL + col] = acc[n][r] + bc;
        }
    }
}

// ---------------------------------------------------------------------------
extern "C" void kernel_launch(void* const* d_in, const int* in_sizes, int n_in,
                              void* d_out, int out_size, void* d_ws, size_t ws_size,
                              hipStream_t stream) {
    const float*         C       = (const float*)d_in[0];
    const unsigned char* nmask   = (const unsigned char*)d_in[1];   // jnp bool: 1 byte
    const float*         centers = (const float*)d_in[2];
    const float*         gamma   = (const float*)d_in[3];
    const float*         beta    = (const float*)d_in[4];
    const float*         W       = (const float*)d_in[5];
    const float*         bias    = (const float*)d_in[6];

    // outputs concatenated: E (f32, NEDGE*128), Kidx (i32, NEDGE), edge_mask (bool, NEDGE)
    char* out = (char*)d_out;
    float*         Eout  = (float*)out;
    int*           Kidx  = (int*)(out + (size_t)NEDGE * DMODEL * sizeof(float));
    unsigned char* Emask = (unsigned char*)(out + (size_t)NEDGE * DMODEL * sizeof(float)
                                                + (size_t)NEDGE * sizeof(int));

    _Float16* Wf = (_Float16*)d_ws;                  // 64 KB fragment-ordered W

    prep_w_kernel<<<128, 256, 0, stream>>>(W, Wf);
    knn_kernel<<<ZB*NN, 256, 0, stream>>>(C, nmask, Kidx, Emask);
    edge_kernel<<<NEDGE/16/8, 256, 0, stream>>>(C, Kidx, centers, gamma, beta, bias, Wf, Eout);
}
